// ReverbNetwork_19722489823829
// MI455X (gfx1250) — matrix operand order repaired and checked by reading, not compiled, not run on hardware
//
#include <hip/hip_runtime.h>

// ---------------------------------------------------------------------------
// ReverbNetwork one-step message passing for MI455X (gfx1250).
//
// new_state[v] = mean_{u in preds(v)} ( conv4x4_SAME(sigmoid(state[u]), W_uv) + b_uv )
//
// Per destination node the mean over predecessors is ONE dense implicit GEMM
// (preds live in the reduction dim):  D[16px x 16couts] = A[16px x K] x B[K x 16]
// with K = preds*Cin*16 (144 or 192), on v_wmma_f32_16x16x32_bf16.
//
// Roofline: ~80 MB HBM @ 23.3 TB/s ~ 3.4 us floor vs only 2.8 GFLOP useful:
// strongly memory-bound, so bf16-in / f32-accumulate WMMA has >100x headroom
// even at 3/16 N-dim utilization. The kernel is specialized on the pred count
// P (3 or 4) so ALL fragment index math folds to compile-time immediates:
// the A build is one v_add per tile + pure ds_load_u16 with immediate offsets
// (PK is a multiple of 16 => K-padding is uniform per 16-wide half = lane
// half; WMMA is outside the pad branch => EXEC stays all-ones around every WMMA).
// ---------------------------------------------------------------------------

#define N_REAL  12
#define BATCH   4
#define CH      3
#define HH      256
#define WW      256
#define KK      4
#define ROWS    4              // output rows per workgroup
#define IN_ROWS (ROWS + 3)     // 7 input rows (pad_low=1, pad_high=2 for K=4 SAME)
#define LCOLS   260            // staged cols: global x in [-1, 258]
#define LSTRIDE 264            // padded LDS row stride
#define MAXP    4              // node 0 has 3 ring preds + driver
#define NWAVES  8              // 256 threads, wave32

typedef __attribute__((ext_vector_type(16))) __bf16 v16bf;
typedef __attribute__((ext_vector_type(8)))  float  v8f;

static __device__ __forceinline__ unsigned short f32_to_bf16(float f) {
  unsigned int x = __builtin_bit_cast(unsigned int, f);
  x += 0x7FFFu + ((x >> 16) & 1u);            // round-to-nearest-even
  return (unsigned short)(x >> 16);
}
static __device__ __forceinline__ __bf16 bf_from_bits(unsigned short u) {
  union { unsigned short u; __bf16 b; } cv; cv.u = u; return cv.b;
}

// ---------------------------------------------------------------------------
// Compute stage, specialized on predecessor count P so PK / chunk count and
// all per-element K decompositions are compile-time.
// ---------------------------------------------------------------------------
template <int P>
static __device__ __forceinline__ void reverb_compute(
    const float* __restrict__ weights,
    const int*   s_pedge,
    const unsigned short (&s_act)[MAXP][CH][IN_ROWS][LSTRIDE],
    float (&s_out)[ROWS][CH][WW],
    int tid)
{
  constexpr int PK     = P * CH * KK * KK;       // 144 or 192
  constexpr int NCHUNK = (PK + 31) / 32;         // 5 or 6

  const int lane  = tid & 31;
  const int h     = lane >> 4;                   // lane half (0/1)
  const int ncol  = lane & 15;                   // N index: cout, valid 0..2
  const int wv    = tid >> 5;                    // wave id
  const int xlane = lane & 15;                   // M index: pixel in 16-wide tile

  // ---- B fragments (weights) in registers, built once ----
  // B layout (16-bit KxN, contiguous k order used here): lane = N; lanes 0-15 hold chunk
  // K=0..15 (elem i -> K = h*16 + i), lanes 16-31 hold K=16..31.
  v16bf bw[NCHUNK];
  const int ncc = (ncol < CH) ? ncol : 0;        // clamped cout for safe loads
  #pragma unroll
  for (int c = 0; c < NCHUNK; ++c) {
    const int  chn   = c * 2 + h;                // (pred, cin) plane; K>>4
    const bool valid = (ncol < CH) && (chn < P * CH);
    const int  chc   = valid ? chn : 0;
    const int  p     = chc / CH;
    const int  cin   = chc - CH * p;
    const int  e     = s_pedge[p];
    const int  wbase = ((e * CH + ncc) * CH + cin) * (KK * KK);
    #pragma unroll
    for (int i = 0; i < 16; ++i) {               // off = ky*4+kx = i (compile-time)
      unsigned short bits = f32_to_bf16(weights[wbase + i]);
      bw[c][i] = bf_from_bits(valid ? bits : (unsigned short)0);
    }
  }

  // ---- 64 output tiles (4 rows x 16 col-tiles), 8 per wave ----
  const unsigned short* actflat = &s_act[0][0][0][0];
  #pragma unroll
  for (int tt = 0; tt < (ROWS * (WW / 16)) / NWAVES; ++tt) {
    const int t  = wv + tt * NWAVES;
    const int r  = t >> 4;                       // output row in tile
    const int xt = t & 15;                       // 16-pixel column tile
    // A layout (matches B): M = lane&15, elem i ->
    // K = h*16 + i.  With chn=K>>4=2c+h, ky=i>>2,
    // kx=i&3, the LDS address splits into (per-tile lane base) + (immediate).
    const int abase = r * LSTRIDE + xt * 16 + xlane;

    v8f acc = {0.f, 0.f, 0.f, 0.f, 0.f, 0.f, 0.f, 0.f};
    #pragma unroll
    for (int c = 0; c < NCHUNK; ++c) {
      v16bf a;
      #pragma unroll
      for (int i = 0; i < 16; ++i) {
        const int g = h;                         // K-half within chunk
        const int j = i;
        if (c * 32 + g * 16 < PK) {              // uniform per lane half (PK % 16 == 0)
          const int chn  = c * 2 + g;            // (pred, cin) plane of this lane half
          const int cofs = (chn * IN_ROWS + (j >> 2)) * LSTRIDE + (j & 3);
          a[i] = bf_from_bits(actflat[abase + cofs]);   // ds_load_u16 +imm
        } else {
          a[i] = bf_from_bits((unsigned short)0);       // K padding
        }
      }
      acc = __builtin_amdgcn_wmma_f32_16x16x32_bf16(
              /*neg_a=*/false, a, /*neg_b=*/false, bw[c],
              /*c_mod=*/(short)0, acc, /*reuse_a=*/false, /*reuse_b=*/false);
    }

    // D layout: lane -> N = lane&15 (cout), elem vi -> M = vi + 8*h (pixel)
    if (ncol < CH) {
      #pragma unroll
      for (int vi = 0; vi < 8; ++vi) {
        s_out[r][ncol][xt * 16 + vi + 8 * h] = acc[vi];
      }
    }
  }
}

__global__ __launch_bounds__(256)
void reverb_step_kernel(const float* __restrict__ states,
                        const float* __restrict__ weights,
                        const float* __restrict__ bias,
                        const int*   __restrict__ edge_src,
                        const int*   __restrict__ edge_dst,
                        float*       __restrict__ out,
                        int E) {
  __shared__ unsigned short s_act[MAXP][CH][IN_ROWS][LSTRIDE]; // bf16 sigmoid tile
  __shared__ float s_out[ROWS][CH][WW];                        // f32 output staging
  __shared__ int   s_psrc[MAXP];
  __shared__ int   s_pedge[MAXP];
  __shared__ int   s_deg;
  __shared__ float s_bsum[CH];

  const int tid = threadIdx.x;
  const int blk = blockIdx.x;
  const int v   = blk / (BATCH * (HH / ROWS));
  const int rb  = blk % (BATCH * (HH / ROWS));
  const int b   = rb / (HH / ROWS);
  const int y0  = (rb % (HH / ROWS)) * ROWS;

  // -------- predecessor list, degree, bias sums (input-driven) --------
  if (tid == 0) {
    int n = 0; float b0 = 0.f, b1 = 0.f, b2 = 0.f;
    for (int e = 0; e < E; ++e) {
      if (edge_dst[e] == v) {
        if (n < MAXP) { s_psrc[n] = edge_src[e]; s_pedge[n] = e; }
        b0 += bias[e*CH + 0]; b1 += bias[e*CH + 1]; b2 += bias[e*CH + 2];
        ++n;
      }
    }
    s_deg = n; s_bsum[0] = b0; s_bsum[1] = b1; s_bsum[2] = b2;
  }
  __syncthreads();

  const int P = s_deg;                  // 3, or 4 for node 0

  // -------- stage sigmoid(state) with zero halo into LDS (bf16) --------
  // Conv pads AFTER sigmoid, so out-of-range halo must be 0.0 (not sigmoid(0)).
  const int tot = P * CH * IN_ROWS * LCOLS;
  for (int idx = tid; idx < tot; idx += 256) {
    int col = idx % LCOLS;
    int t1  = idx / LCOLS;
    int r   = t1 % IN_ROWS;
    int t2  = t1 / IN_ROWS;
    int cin = t2 % CH;
    int p   = t2 / CH;
    int gx = col - 1;                   // pad_low = 1
    int gy = y0 - 1 + r;
    float val = 0.f;
    if (gx >= 0 && gx < WW && gy >= 0 && gy < HH) {
      float x = states[(((s_psrc[p]*BATCH + b)*CH + cin)*HH + gy)*WW + gx];
      val = 1.0f / (1.0f + __expf(-x)); // sigmoid
    }
    s_act[p][cin][r][col] = f32_to_bf16(val);
  }

  // -------- prefetch next row-tile into L2 (global_prefetch_b8) --------
  {
    int p   = tid >> 6;
    int cin = (tid >> 4) & 3;
    int r   = tid & 15;
    int gy  = y0 + ROWS - 1 + r;
    if (p < P && cin < CH && r < IN_ROWS && gy < HH) {
      __builtin_prefetch(&states[(((s_psrc[p]*BATCH + b)*CH + cin)*HH + gy)*WW], 0, 0);
    }
  }
  __syncthreads();

  // -------- WMMA compute, specialized on P (scalar dispatch) --------
  const int Pu = __builtin_amdgcn_readfirstlane(P);
  if (Pu == 4) {
    reverb_compute<4>(weights, s_pedge, s_act, s_out, tid);
  } else {
    reverb_compute<3>(weights, s_pedge, s_act, s_out, tid);
  }
  __syncthreads();

  // -------- coalesced store: (sum + sum_bias) / deg --------
  const float invdeg = 1.0f / (float)P;
  for (int idx = tid; idx < ROWS * CH * WW; idx += 256) {
    int x = idx & (WW - 1);
    int c = (idx >> 8) % CH;
    int r = idx / (CH * WW);
    float val = (s_out[r][c][x] + s_bsum[c]) * invdeg;
    out[(((v*BATCH + b)*CH + c)*HH + (y0 + r))*WW + x] = val;
  }
}

extern "C" void kernel_launch(void* const* d_in, const int* in_sizes, int n_in,
                              void* d_out, int out_size, void* d_ws, size_t ws_size,
                              hipStream_t stream) {
  const float* states  = (const float*)d_in[0];
  const float* weights = (const float*)d_in[1];
  const float* bias    = (const float*)d_in[2];
  const int*   esrc    = (const int*)d_in[3];
  const int*   edst    = (const int*)d_in[4];
  float*       out     = (float*)d_out;
  const int E = in_sizes[3];            // 37 edges

  dim3 grid(N_REAL * BATCH * (HH / ROWS));   // 12 * 4 * 64 = 3072 workgroups
  reverb_step_kernel<<<grid, 256, 0, stream>>>(states, weights, bias, esrc, edst, out, E);
}
